// FullRankRNN_58145267253492
// MI455X (gfx1250) — matrix operand order repaired and checked by reading, not compiled, not run on hardware
//
#include <hip/hip_runtime.h>
#include <hip/hip_bf16.h>
#include <math.h>

typedef __attribute__((ext_vector_type(16))) __bf16 v16bf;
typedef __attribute__((ext_vector_type(8)))  __bf16 v8bf;
typedef __attribute__((ext_vector_type(8)))  float  v8f;

#define B_  64
#define T_  512
#define I_  128
#define H_  1024
#define O_  64
#define NWG 64                 // persistent workgroups (one column-tile each)
#define NOISE_STD 0.05f
#define ALPHA     0.2f

// ---- workspace layout (bf16 elements) ----
#define WIB_OFF   1048576
#define WOB_OFF   1179648
#define R0_OFF    1245184
#define RDELTA    65536        // r1 = r0 + RDELTA (contiguous -> offset ping-pong)
#define BAR_BYTE  2752512      // 2 x u32 barrier state after the bf16 region

static __device__ __forceinline__ v8f zero8() {
  v8f z;
#pragma unroll
  for (int i = 0; i < 8; ++i) z[i] = 0.0f;
  return z;
}

// A-matrix (16x32 bf16) fragment from a row-major [*,1024] bf16 buffer.
// lanes 0-15: K chunks {0..7, 16..23}; lanes 16-31: {8..15, 24..31}; M = lane%16.
static __device__ __forceinline__ v16bf load_a(const __bf16* __restrict__ rm,
                                               int mt, int kt, int lane) {
  const __bf16* p = rm + (mt * 16 + (lane & 15)) * H_ + kt * 32 + ((lane & 16) ? 8 : 0);
  v8bf lo = *(const v8bf*)(p);
  v8bf hi = *(const v8bf*)(p + 16);
  return __builtin_shufflevector(lo, hi, 0,1,2,3,4,5,6,7,8,9,10,11,12,13,14,15);
}

// device-wide sense/generation barrier (all NWG blocks co-resident)
static __device__ __forceinline__ void grid_barrier(unsigned* __restrict__ cnt,
                                                    unsigned* __restrict__ gen,
                                                    unsigned target) {
  __syncthreads();
  if (threadIdx.x == 0) {
    __threadfence();
    unsigned prev = __hip_atomic_fetch_add(cnt, 1u, __ATOMIC_ACQ_REL,
                                           __HIP_MEMORY_SCOPE_AGENT);
    if (prev == NWG - 1) {
      __hip_atomic_store(cnt, 0u, __ATOMIC_RELAXED, __HIP_MEMORY_SCOPE_AGENT);
      __hip_atomic_store(gen, target, __ATOMIC_RELEASE, __HIP_MEMORY_SCOPE_AGENT);
    } else {
      while (__hip_atomic_load(gen, __ATOMIC_ACQUIRE,
                               __HIP_MEMORY_SCOPE_AGENT) != target) {
        __builtin_amdgcn_s_sleep(2);
      }
    }
    __threadfence();
  }
  __syncthreads();
}

// ---------- pack kernels: fp32 weights -> bf16 in WMMA B-matrix lane order ----------
// B-layout (32x16): lane l, elem j -> K = (l<16?0:8)+(j<8?j:j+8), N = l%16; 32B contiguous per lane.
__global__ void pack_wrec_kernel(const float* __restrict__ wrec, __bf16* __restrict__ dst) {
  int idx = blockIdx.x * blockDim.x + threadIdx.x;            // 1,048,576 total
  const int j  = idx & 15;
  const int l  = (idx >> 4) & 31;
  const int nt = (idx >> 9) & 63;
  const int kt = idx >> 15;
  const int N = nt * 16 + (l & 15);                            // output column (h')
  const int K = kt * 32 + ((l & 16) ? 8 : 0) + (j < 8 ? j : j + 8); // reduction (h)
  dst[idx] = (__bf16)wrec[N * H_ + K];                         // wrec^T[K,N] = wrec[N,K]
}

__global__ void pack_wi_kernel(const float* __restrict__ wi, const float* __restrict__ si,
                               __bf16* __restrict__ dst) {
  int idx = blockIdx.x * blockDim.x + threadIdx.x;            // 131,072 total
  const int j  = idx & 15;
  const int l  = (idx >> 4) & 31;
  const int nt = (idx >> 9) & 63;
  const int kt = idx >> 15;                                    // 0..3
  const int N = nt * 16 + (l & 15);                            // h
  const int K = kt * 32 + ((l & 16) ? 8 : 0) + (j < 8 ? j : j + 8); // i
  dst[idx] = (__bf16)(si[K] * wi[K * H_ + N]);
}

__global__ void pack_wo_kernel(const float* __restrict__ wo, const float* __restrict__ so,
                               __bf16* __restrict__ dst, unsigned* __restrict__ bar) {
  int idx = blockIdx.x * blockDim.x + threadIdx.x;            // 65,536 total
  const int j  = idx & 15;
  const int l  = (idx >> 4) & 31;
  const int nt = (idx >> 9) & 3;
  const int kt = idx >> 11;                                    // 0..31
  const int N = nt * 16 + (l & 15);                            // o
  const int K = kt * 32 + ((l & 16) ? 8 : 0) + (j < 8 ? j : j + 8); // h
  dst[idx] = (__bf16)(wo[K * O_ + N] * so[N]);
  if (idx == 0) { bar[0] = 0u; bar[1] = 0u; }                  // reset grid barrier
}

// ---------- persistent RNN scan kernel ----------
__global__ __launch_bounds__(128, 1)
void rnn_scan_kernel(const float* __restrict__ input, const float* __restrict__ noise,
                     const float* __restrict__ h0,
                     const __bf16* __restrict__ wrecB, const __bf16* __restrict__ wiB,
                     const __bf16* __restrict__ woB,
                     __bf16* __restrict__ r0buf,       // r1 = r0buf + RDELTA
                     float* __restrict__ out, float* __restrict__ traj,
                     unsigned* __restrict__ bar) {
  // WG-resident weight slices: loaded from L2 ONCE, reused for all 512 steps.
  __shared__ __align__(32) __bf16 ldsW [32 * 512];   // 32 KB: wrec^T column slice
  __shared__ __align__(32) __bf16 ldsWi[ 4 * 512];   //  4 KB: si*wi column slice
  __shared__ __align__(32) __bf16 ldsWo[32 * 512];   // 32 KB: wo*so slice (readout blocks)

  const int lane = threadIdx.x & 31;
  const int mt   = threadIdx.x >> 5;          // wave index == batch tile (0..3)
  const int nt   = blockIdx.x;                // hidden-column tile (0..63)
  const int col  = nt * 16 + (lane & 15);     // this lane's h column
  const int rhalf = (lane & 16) ? 8 : 0;      // C-layout row offset

  const bool doOut = (mt == 0) && (blockIdx.x < 16);
  const int omt  = blockIdx.x & 3;
  const int ont  = blockIdx.x >> 2;
  const int ocol = ont * 16 + (lane & 15);

  // ---- stage B-matrices into LDS (cooperative, 16B chunks) ----
  for (int i = threadIdx.x; i < 2048; i += 128) {       // wrec: 32 tiles * 512
    const int kt = i >> 6, e = (i & 63) * 8;
    *(v8bf*)&ldsW[kt * 512 + e] = *(const v8bf*)&wrecB[((kt * 64) + nt) * 512 + e];
  }
  for (int i = threadIdx.x; i < 256; i += 128) {        // wi: 4 tiles * 512
    const int kt = i >> 6, e = (i & 63) * 8;
    *(v8bf*)&ldsWi[kt * 512 + e] = *(const v8bf*)&wiB[((kt * 64) + nt) * 512 + e];
  }
  if (blockIdx.x < 16) {                                // wo: 32 tiles * 512
    for (int i = threadIdx.x; i < 2048; i += 128) {
      const int kt = i >> 6, e = (i & 63) * 8;
      *(v8bf*)&ldsWo[kt * 512 + e] = *(const v8bf*)&woB[((kt * 4) + ont) * 512 + e];
    }
  }

  // persistent h tile in C-register layout: lane holds column `col`, rows mt*16+rhalf+v
  float hacc[8];
  const float h0v = h0[col];
  const __bf16 rinit = (__bf16)tanhf(h0v);
#pragma unroll
  for (int v = 0; v < 8; ++v) {
    hacc[v] = h0v;
    r0buf[RDELTA + (mt * 16 + rhalf + v) * H_ + col] = rinit;  // r_{-1} -> buf[1]
  }

  unsigned barGen = 1;
  grid_barrier(bar, bar + 1, barGen++);       // r_{-1} + LDS staging visible

  for (int t = 0; t < T_; ++t) {
    // ping-pong via offset arithmetic on ONE global pointer (keeps global addrspace)
    const __bf16* rin = r0buf + (size_t)((t + 1) & 1) * RDELTA;   // r_{t-1}
    __bf16* rout      = r0buf + (size_t)(t & 1) * RDELTA;         // r_t

    // ---- readout for step t-1 (overlaps with recurrent GEMM of step t) ----
    if (doOut && t > 0) {
      v8f oa0 = zero8(), oa1 = zero8();
#pragma unroll 2
      for (int kt = 0; kt < 32; kt += 2) {
        v16bf a0 = load_a(rin, omt, kt, lane);
        v16bf b0 = *(const v16bf*)&ldsWo[kt * 512 + lane * 16];
        oa0 = __builtin_amdgcn_wmma_f32_16x16x32_bf16(false, a0, false, b0,
                                                      (short)0, oa0, false, false);
        v16bf a1 = load_a(rin, omt, kt + 1, lane);
        v16bf b1 = *(const v16bf*)&ldsWo[(kt + 1) * 512 + lane * 16];
        oa1 = __builtin_amdgcn_wmma_f32_16x16x32_bf16(false, a1, false, b1,
                                                      (short)0, oa1, false, false);
      }
#pragma unroll
      for (int v = 0; v < 8; ++v)
        __builtin_nontemporal_store(oa0[v] + oa1[v],
            &out[((size_t)(omt * 16 + rhalf + v) * T_ + (t - 1)) * O_ + ocol]);
    }

    // ---- recurrent GEMM: acc = r_{t-1} @ wrec^T (tile), K = 1024, dual chains ----
    v8f acc0 = zero8(), acc1 = zero8();
#pragma unroll 2
    for (int kt = 0; kt < 32; kt += 2) {
      v16bf a0 = load_a(rin, mt, kt, lane);
      v16bf b0 = *(const v16bf*)&ldsW[kt * 512 + lane * 16];
      acc0 = __builtin_amdgcn_wmma_f32_16x16x32_bf16(false, a0, false, b0,
                                                     (short)0, acc0, false, false);
      v16bf a1 = load_a(rin, mt, kt + 1, lane);
      v16bf b1 = *(const v16bf*)&ldsW[(kt + 1) * 512 + lane * 16];
      acc1 = __builtin_amdgcn_wmma_f32_16x16x32_bf16(false, a1, false, b1,
                                                     (short)0, acc1, false, false);
    }
    // ---- fused input drive: acc += input_t @ (si*wi) (tile), K = 128 ----
#pragma unroll
    for (int kt = 0; kt < 4; ++kt) {
      const float* ip = input + ((size_t)(mt * 16 + (lane & 15)) * T_ + t) * I_
                        + kt * 32 + ((lane & 16) ? 8 : 0);
      v8f f0 = __builtin_nontemporal_load((const v8f*)(ip));       // read-once stream
      v8f f1 = __builtin_nontemporal_load((const v8f*)(ip + 16));
      v16bf a;
#pragma unroll
      for (int i = 0; i < 8; ++i) { a[i] = (__bf16)f0[i]; a[i + 8] = (__bf16)f1[i]; }
      v16bf b = *(const v16bf*)&ldsWi[kt * 512 + lane * 16];
      if (kt & 1)
        acc1 = __builtin_amdgcn_wmma_f32_16x16x32_bf16(false, a, false, b,
                                                       (short)0, acc1, false, false);
      else
        acc0 = __builtin_amdgcn_wmma_f32_16x16x32_bf16(false, a, false, b,
                                                       (short)0, acc0, false, false);
    }

    // ---- state update, trajectory store, r_t = tanh(h_t) ----
#pragma unroll
    for (int v = 0; v < 8; ++v) {
      const int brow = mt * 16 + rhalf + v;
      const float nz =
          __builtin_nontemporal_load(&noise[((size_t)brow * T_ + t) * H_ + col]);
      float h = hacc[v];
      h = h + NOISE_STD * nz + ALPHA * ((acc0[v] + acc1[v]) - h);
      hacc[v] = h;
      __builtin_nontemporal_store(h, &traj[((size_t)brow * T_ + t) * H_ + col]);
      rout[brow * H_ + col] = (__bf16)tanhf(h);   // reused next step: keep temporal
    }

    grid_barrier(bar, bar + 1, barGen++);     // r_t visible for step t+1
  }

  // ---- readout for the last step (r_511 lives in buf[1]) ----
  if (doOut) {
    const __bf16* rin = r0buf + RDELTA;
    v8f oa0 = zero8(), oa1 = zero8();
#pragma unroll 2
    for (int kt = 0; kt < 32; kt += 2) {
      v16bf a0 = load_a(rin, omt, kt, lane);
      v16bf b0 = *(const v16bf*)&ldsWo[kt * 512 + lane * 16];
      oa0 = __builtin_amdgcn_wmma_f32_16x16x32_bf16(false, a0, false, b0,
                                                    (short)0, oa0, false, false);
      v16bf a1 = load_a(rin, omt, kt + 1, lane);
      v16bf b1 = *(const v16bf*)&ldsWo[(kt + 1) * 512 + lane * 16];
      oa1 = __builtin_amdgcn_wmma_f32_16x16x32_bf16(false, a1, false, b1,
                                                    (short)0, oa1, false, false);
    }
#pragma unroll
    for (int v = 0; v < 8; ++v)
      __builtin_nontemporal_store(oa0[v] + oa1[v],
          &out[((size_t)(omt * 16 + rhalf + v) * T_ + (T_ - 1)) * O_ + ocol]);
  }
}

extern "C" void kernel_launch(void* const* d_in, const int* in_sizes, int n_in,
                              void* d_out, int out_size, void* d_ws, size_t ws_size,
                              hipStream_t stream) {
  (void)in_sizes; (void)n_in; (void)out_size; (void)ws_size;
  const float* input = (const float*)d_in[0];   // (B,T,I)
  const float* noise = (const float*)d_in[1];   // (B,T,H)
  const float* wi    = (const float*)d_in[2];   // (I,H)
  const float* si    = (const float*)d_in[3];   // (I,)
  const float* wrec  = (const float*)d_in[4];   // (H,H)
  const float* wo    = (const float*)d_in[5];   // (H,O)
  const float* so    = (const float*)d_in[6];   // (O,)
  const float* h0    = (const float*)d_in[7];   // (H,)

  float* out  = (float*)d_out;                          // (B,T,O)
  float* traj = out + (size_t)B_ * T_ * O_;             // (B,T,H)

  __bf16* base  = (__bf16*)d_ws;
  __bf16* wrecB = base;
  __bf16* wiB   = base + WIB_OFF;
  __bf16* woB   = base + WOB_OFF;
  __bf16* r0    = base + R0_OFF;
  unsigned* bar = (unsigned*)((char*)d_ws + BAR_BYTE);

  pack_wrec_kernel<<<4096, 256, 0, stream>>>(wrec, wrecB);
  pack_wi_kernel<<<512, 256, 0, stream>>>(wi, si, wiB);
  pack_wo_kernel<<<256, 256, 0, stream>>>(wo, so, woB, bar);
  rnn_scan_kernel<<<NWG, 128, 0, stream>>>(input, noise, h0, wrecB, wiB, woB,
                                           r0, out, traj, bar);
}
